// TemporalFrequencyMasking_80272938763039
// MI455X (gfx1250) — compile-verified, matrix operand order checked by reading, not run on hardware
//
#include <hip/hip_runtime.h>
#include <math.h>

// Problem constants (from reference)
#define BB     2
#define TT     1024
#define DD     263          // d_model
#define CC     526          // N*F
#define MM     (BB*TT)      // 2048 rows
#define WIN    25
#define KT     102
#define KF     102
#define NFREQ  513          // T/2+1
// padded GEMM dims
#define KP_C   544          // pad32(526)
#define KP_D   288          // pad32(263)
#define NP_D   272          // pad16(263)
#define NP_C   528          // pad16(526)

typedef __attribute__((ext_vector_type(16))) _Float16 v16h;
typedef __attribute__((ext_vector_type(8)))  _Float16 v8h;
typedef __attribute__((ext_vector_type(8)))  float    v8f;

// ---------------------------------------------------------------------------
// One wave32 computes one 16x16 f32 tile from pre-packed, zero-padded f16
// operands. A: row-major [M x Kp]; B: column-major [Np x Kp] (column n
// contiguous in k). Fragment order matches the ISA 16-bit A/B 16x32 layout:
// lane<16 holds K j..j+7 at k0+0 and k0+16; lane>=16 at k0+8 and k0+24.
// Inner loop: 4x global_load_b128 + v_wmma_f32_16x16x32_f16, no branches.
// ---------------------------------------------------------------------------
__device__ __forceinline__ v8f wmma_tile_h(const _Float16* __restrict__ A, int lda,
                                           const _Float16* __restrict__ B, int ldb,
                                           int m0, int n0, int Kp) {
  const int lane = threadIdx.x & 31;
  const int l = lane & 15, hs = lane >> 4;
  const _Float16* ap = A + (size_t)(m0 + l) * lda + hs * 8;
  const _Float16* bp = B + (size_t)(n0 + l) * ldb + hs * 8;
  v8f acc = {};
  for (int k0 = 0; k0 < Kp; k0 += 32) {
    v8h a0 = *(const v8h*)(ap + k0);
    v8h a1 = *(const v8h*)(ap + k0 + 16);
    v8h b0 = *(const v8h*)(bp + k0);
    v8h b1 = *(const v8h*)(bp + k0 + 16);
    v16h a = __builtin_shufflevector(a0, a1, 0,1,2,3,4,5,6,7,8,9,10,11,12,13,14,15);
    v16h b = __builtin_shufflevector(b0, b1, 0,1,2,3,4,5,6,7,8,9,10,11,12,13,14,15);
    acc = __builtin_amdgcn_wmma_f32_16x16x32_f16(false, a, false, b, (short)0, acc,
                                                 false, false);
  }
  return acc;
}

// 8 waves per block, one 16x16 tile per wave. tile -> (tm, tn).
#define TILE_PROLOG(tilesN, totalTiles)                              \
  int tile = blockIdx.x * (blockDim.x >> 5) + (threadIdx.x >> 5);    \
  if (tile >= (totalTiles)) return;                                  \
  int tn = tile % (tilesN), tm = tile / (tilesN);                    \
  int n0 = tn * 16, m0 = tm * 16;

#define TILE_EPILOG_VARS()                                           \
  const int lane = threadIdx.x & 31;                                 \
  const int l = lane & 15, hs = lane >> 4;                           \
  const int n = n0 + l;

// --------------------------- operand packing -------------------------------

// out[n*Kp+k] = W[n*C+k]  (zero-padded). Packs W^T-as-B and A matrices.
__global__ void k_pack_T(const float* __restrict__ W, _Float16* __restrict__ out,
                         int R, int C, int Np, int Kp) {
  size_t i = (size_t)blockIdx.x * blockDim.x + threadIdx.x;
  if (i >= (size_t)Np * Kp) return;
  int n = (int)(i / Kp), k = (int)(i % Kp);
  float v = (n < R && k < C) ? W[(size_t)n * C + k] : 0.f;
  out[i] = (_Float16)v;
}

// out[n*Kp+k] = W[k*C+n]  (row-major B, zero-padded transpose copy)
__global__ void k_pack_B(const float* __restrict__ W, _Float16* __restrict__ out,
                         int R, int C, int Np, int Kp) {
  size_t i = (size_t)blockIdx.x * blockDim.x + threadIdx.x;
  if (i >= (size_t)Np * Kp) return;
  int n = (int)(i / Kp), k = (int)(i % Kp);
  float v = (k < R && n < C) ? W[(size_t)k * C + n] : 0.f;
  out[i] = (_Float16)v;
}

__global__ void k_zero16(_Float16* __restrict__ p, size_t n) {
  size_t i = (size_t)blockIdx.x * blockDim.x + threadIdx.x;
  if (i < n) p[i] = (_Float16)0.f;
}

__global__ void k_zero(float* __restrict__ p, int n) {
  int i = blockIdx.x * blockDim.x + threadIdx.x;
  if (i < n) p[i] = 0.f;
}

// ------------------------- GEMM kernels with epilogues ---------------------

// ex = X @ W_emb^T + b_emb + PE (f32 out; feeds score/FFT/masking)
__global__ void k_gemm_embed(const _Float16* __restrict__ Ah, const _Float16* __restrict__ Bh,
                             const float* __restrict__ bias, float* __restrict__ ex,
                             int tilesN, int totalTiles) {
  TILE_PROLOG(tilesN, totalTiles);
  v8f acc = wmma_tile_h(Ah, KP_C, Bh, KP_C, m0, n0, KP_C);
  TILE_EPILOG_VARS();
  if (n >= DD) return;
  float e = expf(-9.2103404f * (float)(n & ~1) / (float)DD);  // 10000^{-(2i)/D}
  float bn = bias[n];
#pragma unroll
  for (int r = 0; r < 8; ++r) {
    int m = m0 + r + hs * 8;
    int t = m & (TT - 1);
    float pe = (n & 1) ? cosf((float)t * e) : sinf((float)t * e);
    ex[(size_t)m * DD + n] = acc[r] + bn + pe;
  }
}

// h = gelu(masked @ W1t^T + b1t)  -> padded f16 (A of next GEMM)
__global__ void k_gemm_gelu(const _Float16* __restrict__ Ah, const _Float16* __restrict__ Bh,
                            const float* __restrict__ bias, _Float16* __restrict__ outh,
                            int tilesN, int totalTiles) {
  TILE_PROLOG(tilesN, totalTiles);
  v8f acc = wmma_tile_h(Ah, KP_D, Bh, KP_D, m0, n0, KP_D);
  TILE_EPILOG_VARS();
  if (n >= DD) return;
  float bn = bias[n];
#pragma unroll
  for (int r = 0; r < 8; ++r) {
    int m = m0 + r + hs * 8;
    float x = acc[r] + bn;
    outh[(size_t)m * KP_D + n] = (_Float16)(0.5f * x * (1.f + erff(x * 0.70710678f)));
  }
}

// proj = sigmoid(H @ W2t^T + b2t); masked2 = mask_t ? masked : proj -> padded f16
__global__ void k_gemm_sigwhere(const _Float16* __restrict__ Ah, const _Float16* __restrict__ Bh,
                                const float* __restrict__ bias,
                                const float* __restrict__ masked,
                                const float* __restrict__ mskt,
                                _Float16* __restrict__ outh, int tilesN, int totalTiles) {
  TILE_PROLOG(tilesN, totalTiles);
  v8f acc = wmma_tile_h(Ah, KP_D, Bh, KP_D, m0, n0, KP_D);
  TILE_EPILOG_VARS();
  if (n >= DD) return;
  float bn = bias[n];
#pragma unroll
  for (int r = 0; r < 8; ++r) {
    int m = m0 + r + hs * 8;
    float p = 1.f / (1.f + expf(-(acc[r] + bn)));
    float v = (mskt[m] != 0.f) ? masked[(size_t)m * DD + n] : p;
    outh[(size_t)m * KP_D + n] = (_Float16)v;
  }
}

// out = A @ B (f32 store), generic N/Kp — used for out_t and v
__global__ void k_gemm_rowB(const _Float16* __restrict__ Ah, const _Float16* __restrict__ Bh,
                            float* __restrict__ out, int N, int Kp,
                            int tilesN, int totalTiles) {
  TILE_PROLOG(tilesN, totalTiles);
  v8f acc = wmma_tile_h(Ah, Kp, Bh, Kp, m0, n0, Kp);
  TILE_EPILOG_VARS();
  if (n >= N) return;
#pragma unroll
  for (int r = 0; r < 8; ++r) {
    int m = m0 + r + hs * 8;
    out[(size_t)m * N + n] = acc[r];
  }
}

// ------------------------- temporal scoring / top-k ------------------------

__global__ void k_score(const float* __restrict__ ex, float* __restrict__ score) {
  int m = blockIdx.x * blockDim.x + threadIdx.x;
  if (m >= MM) return;
  int b = m >> 10, t = m & (TT - 1);
  int w0 = t - (WIN - 1); if (w0 < 0) w0 = 0;
  float c = (float)(t - w0 + 1);
  float var_sum = 0.f, m1_sum = 0.f;
  for (int d = 0; d < DD; ++d) {
    float s1 = 0.f, s2 = 0.f;
    for (int tt = w0; tt <= t; ++tt) {
      float e = ex[((size_t)b * TT + tt) * DD + d];
      s1 += e; s2 += e * e;
    }
    float mu = s1 / c;
    var_sum += s2 / c - mu * mu;
    m1_sum  += mu;
  }
  score[m] = var_sum / (m1_sum + 1e-6f);
}

// iterative top-k (first-occurrence tie-break like lax.top_k); one block/batch
__global__ void k_topk(const float* __restrict__ score, int n, int k,
                       int* __restrict__ idx_out, float* __restrict__ idxf_out,
                       float* __restrict__ mask, int mask_n) {
  int b = blockIdx.x, tid = threadIdx.x;
  const float* s = score + (size_t)b * n;
  __shared__ float bv[128];
  __shared__ int   bi[128];
  __shared__ unsigned char used[1024];
  for (int i = tid; i < n; i += blockDim.x) used[i] = 0;
  if (mask) for (int i = tid; i < mask_n; i += blockDim.x) mask[(size_t)b * mask_n + i] = 0.f;
  __syncthreads();
  for (int r = 0; r < k; ++r) {
    float best = -INFINITY; int besti = n;
    for (int i = tid; i < n; i += blockDim.x) {
      if (!used[i]) {
        float v = s[i];
        if (v > best || (v == best && i < besti)) { best = v; besti = i; }
      }
    }
    bv[tid] = best; bi[tid] = besti;
    __syncthreads();
    for (int off = 64; off > 0; off >>= 1) {
      if (tid < off) {
        float ov = bv[tid + off]; int oi = bi[tid + off];
        if (ov > bv[tid] || (ov == bv[tid] && oi < bi[tid])) { bv[tid] = ov; bi[tid] = oi; }
      }
      __syncthreads();
    }
    if (tid == 0) {
      int sel = bi[0];
      used[sel] = 1;
      idx_out[(size_t)b * k + r]  = sel;
      idxf_out[(size_t)b * k + r] = (float)sel;
      if (mask) mask[(size_t)b * mask_n + sel] = 1.f;
    }
    __syncthreads();
  }
}

// masked (f32, for sigwhere epilogue) + masked_h (padded f16 A operand)
__global__ void k_build_masked(const float* __restrict__ ex, const float* __restrict__ mskt,
                               const float* __restrict__ token,
                               float* __restrict__ masked, _Float16* __restrict__ maskedh) {
  size_t i = (size_t)blockIdx.x * blockDim.x + threadIdx.x;
  if (i >= (size_t)MM * KP_D) return;
  size_t m = i / KP_D; int k = (int)(i % KP_D);
  float v = 0.f;
  if (k < DD) {
    v = (mskt[m] != 0.f) ? token[k] : ex[m * DD + k];
    masked[m * DD + k] = v;
  }
  maskedh[i] = (_Float16)v;
}

// ------------------------- FFT (radix-2 DIT in LDS) ------------------------

__global__ void k_fft_fwd(const float* __restrict__ ex, float* __restrict__ cre,
                          float* __restrict__ cim, float* __restrict__ magsum) {
  int b = blockIdx.x / DD, d = blockIdx.x % DD;
  __shared__ float sr[TT], si[TT];
  int tid = threadIdx.x;
  for (int t = tid; t < TT; t += blockDim.x) {
    int r = __brev((unsigned)t) >> 22;           // 10-bit bit-reverse
    sr[r] = ex[((size_t)b * TT + t) * DD + d];
    si[r] = 0.f;
  }
  __syncthreads();
  for (int s = 0; s < 10; ++s) {
    int half = 1 << s, len = half << 1;
    for (int i = tid; i < TT / 2; i += blockDim.x) {
      int pos = i & (half - 1);
      int i0 = ((i >> s) << (s + 1)) | pos, i1 = i0 + half;
      float ang = -6.2831853f * (float)pos / (float)len;
      float wr = cosf(ang), wi = sinf(ang);
      float tr = sr[i1] * wr - si[i1] * wi;
      float ti = sr[i1] * wi + si[i1] * wr;
      sr[i1] = sr[i0] - tr; si[i1] = si[i0] - ti;
      sr[i0] += tr;         si[i0] += ti;
    }
    __syncthreads();
  }
  for (int kk = tid; kk < NFREQ; kk += blockDim.x) {
    float re = sr[kk], im = si[kk];
    size_t o = ((size_t)b * NFREQ + kk) * DD + d;
    cre[o] = re; cim[o] = im;
    atomicAdd(&magsum[(size_t)b * NFREQ + kk], sqrtf(re * re + im * im) * (1.f / (float)DD));
  }
}

__global__ void k_scatter_freq(const int* __restrict__ idx_f, const float* __restrict__ tr,
                               const float* __restrict__ ti, float* __restrict__ cre,
                               float* __restrict__ cim) {
  int b = blockIdx.x / KF, j = blockIdx.x % KF;
  int k = idx_f[(size_t)b * KF + j];
  for (int d = threadIdx.x; d < DD; d += blockDim.x) {
    size_t o = ((size_t)b * NFREQ + k) * DD + d;
    cre[o] = tr[d]; cim[o] = ti[d];
  }
}

// inverse FFT; writes xt directly as padded f16 A operand for v = xt @ W_emb
__global__ void k_fft_inv(const float* __restrict__ cre, const float* __restrict__ cim,
                          _Float16* __restrict__ xth) {
  int b = blockIdx.x / DD, d = blockIdx.x % DD;
  __shared__ float sr[TT], si[TT];
  int tid = threadIdx.x;
  for (int kk = tid; kk < TT; kk += blockDim.x) {
    int src = (kk <= TT / 2) ? kk : TT - kk;       // Hermitian extension
    size_t o = ((size_t)b * NFREQ + src) * DD + d;
    float re = cre[o], im = cim[o];
    if (kk > TT / 2) im = -im;
    int r = __brev((unsigned)kk) >> 22;
    sr[r] = re; si[r] = im;
  }
  __syncthreads();
  for (int s = 0; s < 10; ++s) {
    int half = 1 << s, len = half << 1;
    for (int i = tid; i < TT / 2; i += blockDim.x) {
      int pos = i & (half - 1);
      int i0 = ((i >> s) << (s + 1)) | pos, i1 = i0 + half;
      float ang = 6.2831853f * (float)pos / (float)len;   // inverse: +i
      float wr = cosf(ang), wi = sinf(ang);
      float tr = sr[i1] * wr - si[i1] * wi;
      float ti = sr[i1] * wi + si[i1] * wr;
      sr[i1] = sr[i0] - tr; si[i1] = si[i0] - ti;
      sr[i0] += tr;         si[i0] += ti;
    }
    __syncthreads();
  }
  for (int t = tid; t < TT; t += blockDim.x)
    xth[((size_t)b * TT + t) * KP_D + d] = (_Float16)(sr[t] * (1.f / (float)TT));
}

// mask_f = irfft(one-hot spectrum) != 0, evaluated directly from cosine sum
__global__ void k_maskf(const int* __restrict__ idx_f, float* __restrict__ mask_f) {
  int m = blockIdx.x * blockDim.x + threadIdx.x;
  if (m >= MM) return;
  int b = m >> 10, t = m & (TT - 1);
  float acc = 0.f;
  for (int j = 0; j < KF; ++j) {
    int k = idx_f[(size_t)b * KF + j];
    float w;
    if (k == 0)            w = 1.f;
    else if (k == TT / 2)  w = cosf(3.14159265f * (float)t);
    else                   w = 2.f * cosf(6.2831853f * (float)k * (float)t / (float)TT);
    acc += w;
  }
  mask_f[m] = ((acc * (1.f / (float)TT)) != 0.f) ? 1.f : 0.f;
}

// per-scalar MLP with LDS-staged weights (hot kernel: ~283M gelu/FMA)
__global__ void k_mlp_freq(const float* __restrict__ v, const float* __restrict__ w1f,
                           const float* __restrict__ b1f, const float* __restrict__ w2f,
                           const float* __restrict__ b2f, const float* __restrict__ mask_f,
                           float* __restrict__ out_f) {
  __shared__ float sw1[DD], sb1[DD], sw2[DD];
  for (int d = threadIdx.x; d < DD; d += blockDim.x) {
    sw1[d] = w1f[d]; sb1[d] = b1f[d]; sw2[d] = w2f[d];
  }
  __syncthreads();
  size_t i = (size_t)blockIdx.x * blockDim.x + threadIdx.x;
  if (i >= (size_t)MM * CC) return;
  size_t m = i / CC;
  float val = v[i];
  float acc = 0.f;
  for (int d = 0; d < DD; ++d) {
    float hx = val * sw1[d] + sb1[d];
    float g = 0.5f * hx * (1.f + erff(hx * 0.70710678f));
    acc += g * sw2[d];
  }
  float pf = 1.f / (1.f + expf(-(acc + b2f[0])));
  out_f[i] = (mask_f[m] != 0.f) ? val : pf;
}

// ---------------------------------------------------------------------------

extern "C" void kernel_launch(void* const* d_in, const int* in_sizes, int n_in,
                              void* d_out, int out_size, void* d_ws, size_t ws_size,
                              hipStream_t stream) {
  (void)in_sizes; (void)n_in; (void)out_size; (void)ws_size;
  const float* x     = (const float*)d_in[0];
  const float* W_emb = (const float*)d_in[1];
  const float* b_emb = (const float*)d_in[2];
  const float* t_tok = (const float*)d_in[3];
  const float* W1t   = (const float*)d_in[4];
  const float* b1t   = (const float*)d_in[5];
  const float* W2t   = (const float*)d_in[6];
  const float* b2t   = (const float*)d_in[7];
  const float* ftr   = (const float*)d_in[8];
  const float* fti   = (const float*)d_in[9];
  const float* w1f   = (const float*)d_in[10];
  const float* b1f   = (const float*)d_in[11];
  const float* w2f   = (const float*)d_in[12];
  const float* b2f   = (const float*)d_in[13];

  float* outp = (float*)d_out;
  const size_t SZ_OUT = (size_t)MM * CC;
  float* out_t  = outp;
  float* idxt_f = outp + SZ_OUT;
  float* out_f  = outp + SZ_OUT + BB * KT;
  float* idxf_f = outp + 2 * SZ_OUT + BB * KT;

  // ---- workspace layout: f32 region then 32B-aligned f16 arena ----
  float* ws = (float*)d_ws;
  size_t off = 0;
  auto af = [&](size_t n) { float* p = ws + off; off += (n + 7) & ~(size_t)7; return p; };
  float* ex     = af((size_t)MM * DD);
  float* masked = af((size_t)MM * DD);
  float* vbuf   = af((size_t)MM * CC);
  float* cre    = af((size_t)BB * NFREQ * DD);
  float* cim    = af((size_t)BB * NFREQ * DD);
  float* score  = af(MM);
  float* mskt   = af(MM);
  float* maskf  = af(MM);
  float* magsum = af((size_t)BB * NFREQ);
  int*   idx_t  = (int*)af(BB * KT);
  int*   idx_f  = (int*)af(BB * KF);

  _Float16* hbase = (_Float16*)(ws + off);
  size_t hoff = 0;
  auto ah = [&](size_t n) { _Float16* p = hbase + hoff; hoff += (n + 15) & ~(size_t)15; return p; };
  _Float16* X_h      = ah((size_t)MM * KP_C);
  _Float16* WembT_h  = ah((size_t)NP_D * KP_C);
  _Float16* WembB_h  = ah((size_t)NP_C * KP_D);
  _Float16* W1t_h    = ah((size_t)NP_D * KP_D);
  _Float16* W2t_h    = ah((size_t)NP_D * KP_D);
  _Float16* masked_h = ah((size_t)MM * KP_D);
  _Float16* h_h      = ah((size_t)MM * KP_D);   // these three are contiguous:
  _Float16* m2_h     = ah((size_t)MM * KP_D);   // zeroed in one shot below
  _Float16* xt_h     = ah((size_t)MM * KP_D);

  const int tM = MM / 16;                        // 128
  const int tD = NP_D / 16;                      // 17
  const int tC = NP_C / 16;                      // 33
  const int tilesDD = tD * tM;                   // 2176 (divisible by 8)
  const int tilesCC = tC * tM;                   // 4224 (divisible by 8)

  auto g1 = [](size_t n) { return dim3((unsigned)((n + 255) / 256)); };

  // ---- pack f16 operands (pad regions written as zero) ----
  k_pack_T<<<g1((size_t)MM * KP_C), dim3(256), 0, stream>>>(x, X_h, MM, CC, MM, KP_C);
  k_pack_T<<<g1((size_t)NP_D * KP_C), dim3(256), 0, stream>>>(W_emb, WembT_h, DD, CC, NP_D, KP_C);
  k_pack_B<<<g1((size_t)NP_C * KP_D), dim3(256), 0, stream>>>(W_emb, WembB_h, DD, CC, NP_C, KP_D);
  k_pack_T<<<g1((size_t)NP_D * KP_D), dim3(256), 0, stream>>>(W1t, W1t_h, DD, DD, NP_D, KP_D);
  k_pack_T<<<g1((size_t)NP_D * KP_D), dim3(256), 0, stream>>>(W2t, W2t_h, DD, DD, NP_D, KP_D);
  // zero pad columns of the epilogue-written f16 A operands (h, masked2, xt)
  k_zero16<<<g1((size_t)3 * MM * KP_D), dim3(256), 0, stream>>>(h_h, (size_t)3 * MM * KP_D);

  // ---- temporal branch ----
  k_gemm_embed<<<dim3(tilesDD / 8), dim3(256), 0, stream>>>(X_h, WembT_h, b_emb, ex,
                                                            tD, tilesDD);
  k_score<<<g1(MM), dim3(256), 0, stream>>>(ex, score);
  k_topk<<<dim3(BB), dim3(128), 0, stream>>>(score, TT, KT, idx_t, idxt_f, mskt, TT);
  k_build_masked<<<g1((size_t)MM * KP_D), dim3(256), 0, stream>>>(ex, mskt, t_tok,
                                                                  masked, masked_h);
  k_gemm_gelu<<<dim3(tilesDD / 8), dim3(256), 0, stream>>>(masked_h, W1t_h, b1t, h_h,
                                                           tD, tilesDD);
  k_gemm_sigwhere<<<dim3(tilesDD / 8), dim3(256), 0, stream>>>(h_h, W2t_h, b2t, masked,
                                                               mskt, m2_h, tD, tilesDD);
  k_gemm_rowB<<<dim3(tilesCC / 8), dim3(256), 0, stream>>>(m2_h, WembB_h, out_t,
                                                           CC, KP_D, tC, tilesCC);

  // ---- frequency branch ----
  k_zero<<<g1(BB * NFREQ), dim3(256), 0, stream>>>(magsum, BB * NFREQ);
  k_fft_fwd<<<dim3(BB * DD), dim3(128), 0, stream>>>(ex, cre, cim, magsum);
  k_topk<<<dim3(BB), dim3(128), 0, stream>>>(magsum, NFREQ, KF, idx_f, idxf_f,
                                             (float*)nullptr, 0);
  k_scatter_freq<<<dim3(BB * KF), dim3(128), 0, stream>>>(idx_f, ftr, fti, cre, cim);
  k_fft_inv<<<dim3(BB * DD), dim3(128), 0, stream>>>(cre, cim, xt_h);
  k_maskf<<<g1(MM), dim3(256), 0, stream>>>(idx_f, maskf);
  k_gemm_rowB<<<dim3(tilesCC / 8), dim3(256), 0, stream>>>(xt_h, WembB_h, vbuf,
                                                           CC, KP_D, tC, tilesCC);
  k_mlp_freq<<<g1((size_t)MM * CC), dim3(256), 0, stream>>>(vbuf, w1f, b1f, w2f, b2f,
                                                            maskf, out_f);
}
